// AQExpert_11579231830501
// MI455X (gfx1250) — matrix operand order repaired and checked by reading, not compiled, not run on hardware
//
#include <hip/hip_runtime.h>

// ---------------------------------------------------------------------------
// AQLM-style additive-quantized linear:  out = clip((x*scales) @ W^T, +-50)
// Pipeline:
//   (1) dequant kernel folds scales into W  -> bf16 W' in d_ws
//   (2) convert X f32 -> bf16 once          -> Xb in d_ws (if ws big enough)
//   (3) bf16 WMMA GEMM, BK=64, LDS double buffered, staged with CDNA5
//       GLOBAL_LOAD_ASYNC_TO_LDS_B128 (ASYNCcnt) — no VGPR staging.
// Fallback (small ws): f32-X GEMM with in-loop v_cvt_pk_bf16_f32.
// ---------------------------------------------------------------------------

typedef __attribute__((ext_vector_type(16))) __bf16 v16bf;
typedef __attribute__((ext_vector_type(2)))  __bf16 v2bf;
typedef __attribute__((ext_vector_type(8)))  float  v8f;

#define IN_F   4096
#define OUT_F  4096
#define NGRP   (IN_F / 8)     // 512 groups per output row
#define CBSZ   256

#define BM 128
#define BN 128

static __device__ __forceinline__ unsigned int pack_bf16_2(float a, float b) {
    v2bf t; t[0] = (__bf16)a; t[1] = (__bf16)b;     // -> v_cvt_pk_bf16_f32
    return __builtin_bit_cast(unsigned int, t);
}
static __device__ __forceinline__ float bf16_lo(unsigned int w) {
    return __builtin_bit_cast(float, w << 16);
}
static __device__ __forceinline__ float bf16_hi(unsigned int w) {
    return __builtin_bit_cast(float, w & 0xFFFF0000u);
}

// CDNA5 async global->LDS copy (16B per lane), tracked by ASYNCcnt.
static __device__ __forceinline__ void async_load_b128(unsigned lds_off,
                                                       unsigned long long gaddr) {
    asm volatile("global_load_async_to_lds_b128 %0, %1, off"
                 :: "v"(lds_off), "v"(gaddr) : "memory");
}
static __device__ __forceinline__ void wait_async0() {
    asm volatile("s_wait_asynccnt 0x0" ::: "memory");
}

union FragU { v16bf v; uint4 q[2]; };

// ---------------------------------------------------------------------------
// Kernel 1: dequantize + fold scales.  One thread per 8-wide group.
// ---------------------------------------------------------------------------
__global__ __launch_bounds__(256) void aq_dequant_kernel(
    const int* __restrict__ indices,          // [OUT_F, NGRP, 2]
    const unsigned short* __restrict__ cb,    // [2, 256, 8] bf16
    const float* __restrict__ scales,         // [IN_F]
    unsigned short* __restrict__ Wq)          // [OUT_F, IN_F] bf16 out
{
    const int g  = blockIdx.x * 256 + threadIdx.x;
    const int gi = g & (NGRP - 1);

    const int2 id = ((const int2*)indices)[g];
    const uint4 e0 = ((const uint4*)cb)[id.x];
    const uint4 e1 = ((const uint4*)cb)[CBSZ + id.y];

    float s[8];
    *(float4*)&s[0] = ((const float4*)scales)[gi * 2];
    *(float4*)&s[4] = ((const float4*)scales)[gi * 2 + 1];

    const unsigned int* a = (const unsigned int*)&e0;
    const unsigned int* b = (const unsigned int*)&e1;

    float v[8];
#pragma unroll
    for (int j = 0; j < 4; ++j) {
        v[2 * j]     = bf16_lo(a[j]) + bf16_lo(b[j]);
        v[2 * j + 1] = bf16_hi(a[j]) + bf16_hi(b[j]);
    }
#pragma unroll
    for (int j = 0; j < 8; ++j) v[j] *= s[j];

    uint4 o;
    o.x = pack_bf16_2(v[0], v[1]);
    o.y = pack_bf16_2(v[2], v[3]);
    o.z = pack_bf16_2(v[4], v[5]);
    o.w = pack_bf16_2(v[6], v[7]);
    ((uint4*)Wq)[g] = o;
}

// ---------------------------------------------------------------------------
// Kernel 2: X f32 -> bf16 (one pass)
// ---------------------------------------------------------------------------
__global__ __launch_bounds__(256) void aq_xcvt_kernel(
    const float* __restrict__ X, unsigned short* __restrict__ Xb)
{
    const size_t i = (size_t)blockIdx.x * 256 + threadIdx.x;  // per 8 elements
    const float4 a = ((const float4*)X)[2 * i];
    const float4 b = ((const float4*)X)[2 * i + 1];
    uint4 o;
    o.x = pack_bf16_2(a.x, a.y);
    o.y = pack_bf16_2(a.z, a.w);
    o.z = pack_bf16_2(b.x, b.y);
    o.w = pack_bf16_2(b.z, b.w);
    ((uint4*)Xb)[i] = o;
}

// ---------------------------------------------------------------------------
// Kernel 3 (main): bf16 GEMM, async-to-LDS staging.
// 256 threads = 8 waves (4x2); block tile 128x128x64; wave tile 32x64.
// LDS: A/B tiles 128 x 64 bf16, stride 72 halves (conflict-free frag loads).
// ---------------------------------------------------------------------------
#define BK2   64
#define LDK   72
#define T_A0  0
#define T_A1  (BM * LDK)
#define T_B0  (2 * BM * LDK)
#define T_B1  (2 * BM * LDK + BN * LDK)
#define SMEM2 (2 * BM * LDK + 2 * BN * LDK)
#define BUFSTRIDE_BYTES (BM * LDK * 2)     // 18432 B between double buffers

__global__ __launch_bounds__(256) void aq_gemm_bf16_kernel(
    const unsigned short* __restrict__ Xb,
    const unsigned short* __restrict__ Wq,
    float* __restrict__ Out, int M)
{
    __shared__ __align__(16) unsigned short smem[SMEM2];

    const int tid  = threadIdx.x;
    const int lane = tid & 31;
    const int wave = tid >> 5;
    const int wm   = wave >> 1;      // 0..3
    const int wn   = wave & 1;       // 0..1
    const int bm   = blockIdx.y * BM;
    const int bn   = blockIdx.x * BN;

    // Staging geometry: 128 rows x 64 bf16 per tile; 16B chunk per slot.
    unsigned ldsA0[4], ldsB0[4];         // LDS byte addrs, buffer 0
    unsigned long long gA[4], gB[4];     // global byte addrs (advance 128 B/stage)
#pragma unroll
    for (int i = 0; i < 4; ++i) {
        const int li  = tid + i * 256;
        const int row = li >> 3;
        const int col = (li & 7) * 8;
        const unsigned eoff = row * LDK + col;
        ldsA0[i] = (unsigned)(size_t)&smem[T_A0 + eoff];
        ldsB0[i] = (unsigned)(size_t)&smem[T_B0 + eoff];
        gA[i] = (unsigned long long)(size_t)(Xb + (size_t)(bm + row) * IN_F + col);
        gB[i] = (unsigned long long)(size_t)(Wq + (size_t)(bn + row) * IN_F + col);
    }

    v8f acc[2][4] = {};
    const int r16 = lane & 15;
    const int kA  = (lane >> 4) * 8;     // A frag K-chunk select (ISA layout)
    const int kB  = (lane >> 4) * 8;     // B frag K-chunk select (ISA layout)

    // ---- prologue: async-stage tile 0 into buffer 0 ----
#pragma unroll
    for (int i = 0; i < 4; ++i) async_load_b128(ldsA0[i], gA[i]);
#pragma unroll
    for (int i = 0; i < 4; ++i) async_load_b128(ldsB0[i], gB[i]);
#pragma unroll
    for (int i = 0; i < 4; ++i) { gA[i] += BK2 * 2; gB[i] += BK2 * 2; }
    wait_async0();
    __syncthreads();

    const int nk = IN_F / BK2;           // 64 stages
    for (int kt = 0; kt < nk; ++kt) {
        const int buf = kt & 1;
        const unsigned cAoff = buf ? T_A1 : T_A0;
        const unsigned cBoff = buf ? T_B1 : T_B0;

        // Async-stage next tile into the other buffer (no VGPR staging).
        if (kt + 1 < nk) {
            const unsigned sel = buf ? 0u : (unsigned)BUFSTRIDE_BYTES;
#pragma unroll
            for (int i = 0; i < 4; ++i) async_load_b128(ldsA0[i] + sel, gA[i]);
#pragma unroll
            for (int i = 0; i < 4; ++i) async_load_b128(ldsB0[i] + sel, gB[i]);
#pragma unroll
            for (int i = 0; i < 4; ++i) { gA[i] += BK2 * 2; gB[i] += BK2 * 2; }
        }

        // Two 16x16x32 K-steps per stage
#pragma unroll
        for (int ks = 0; ks < 2; ++ks) {
            const int kbase = ks * 32;
            v16bf af[2], bfr[4];
#pragma unroll
            for (int mt = 0; mt < 2; ++mt) {
                const unsigned base = cAoff + (wm * 32 + mt * 16 + r16) * LDK + kbase;
                FragU f;
                f.q[0] = *(const uint4*)&smem[base + kA];
                f.q[1] = *(const uint4*)&smem[base + kA + 16];
                af[mt] = f.v;
            }
#pragma unroll
            for (int nt = 0; nt < 4; ++nt) {
                const unsigned base = cBoff + (wn * 64 + nt * 16 + r16) * LDK + kbase + kB;
                FragU f;
                f.q[0] = *(const uint4*)&smem[base];
                f.q[1] = *(const uint4*)&smem[base + 16];
                bfr[nt] = f.v;
            }
#pragma unroll
            for (int mt = 0; mt < 2; ++mt)
#pragma unroll
                for (int nt = 0; nt < 4; ++nt)
                    acc[mt][nt] = __builtin_amdgcn_wmma_f32_16x16x32_bf16(
                        false, af[mt], false, bfr[nt], (short)0, acc[mt][nt], false, false);
        }

        wait_async0();       // our async writes to buf^1 have landed in LDS
        __syncthreads();     // ... and are visible to all waves
    }

    // ---- epilogue: clamp and store ----
    const int rhi = (lane >> 4) * 8;
#pragma unroll
    for (int mt = 0; mt < 2; ++mt)
#pragma unroll
        for (int nt = 0; nt < 4; ++nt)
#pragma unroll
            for (int r = 0; r < 8; ++r) {
                float v = acc[mt][nt][r];
                v = fminf(50.0f, fmaxf(-50.0f, v));
                const int row = bm + wm * 32 + mt * 16 + rhi + r;
                const int col = bn + wn * 64 + nt * 16 + r16;
                Out[(size_t)row * OUT_F + col] = v;
            }
}

// ---------------------------------------------------------------------------
// Kernel 3b (fallback if ws too small for Xb): f32-X GEMM, in-loop convert.
// ---------------------------------------------------------------------------
#define BK 32
#define LDA 40
#define F_A0 0
#define F_A1 (BM * LDA)
#define F_B0 (2 * BM * LDA)
#define F_B1 (2 * BM * LDA + BN * LDA)
#define SMEM1 (2 * BM * LDA + 2 * BN * LDA)

__global__ __launch_bounds__(256) void aq_gemm_f32_kernel(
    const float* __restrict__ X,
    const unsigned short* __restrict__ Wq,
    float* __restrict__ Out, int M)
{
    __shared__ __align__(16) unsigned short smem[SMEM1];

    const int tid  = threadIdx.x;
    const int lane = tid & 31;
    const int wave = tid >> 5;
    const int wm   = wave >> 1;
    const int wn   = wave & 1;
    const int bm   = blockIdx.y * BM;
    const int bn   = blockIdx.x * BN;

    int aRow[4], aCol[4];
#pragma unroll
    for (int i = 0; i < 4; ++i) { int li = tid + i * 256; aRow[i] = li >> 3; aCol[i] = (li & 7) * 4; }
    int bRow[2], bCol[2];
#pragma unroll
    for (int i = 0; i < 2; ++i) { int li = tid + i * 256; bRow[i] = li >> 2; bCol[i] = (li & 3) * 8; }

    float4 ar[4];
    uint4  br[2];
    v8f acc[2][4] = {};

    const int r16 = lane & 15;
    const int kA  = (lane >> 4) * 8;
    const int kB  = (lane >> 4) * 8;

#pragma unroll
    for (int i = 0; i < 4; ++i)
        ar[i] = *(const float4*)&X[(size_t)(bm + aRow[i]) * IN_F + aCol[i]];
#pragma unroll
    for (int i = 0; i < 2; ++i)
        br[i] = *(const uint4*)&Wq[(size_t)(bn + bRow[i]) * IN_F + bCol[i]];
#pragma unroll
    for (int i = 0; i < 4; ++i) {
        uint2 t; t.x = pack_bf16_2(ar[i].x, ar[i].y); t.y = pack_bf16_2(ar[i].z, ar[i].w);
        *(uint2*)&smem[F_A0 + aRow[i] * LDA + aCol[i]] = t;
    }
#pragma unroll
    for (int i = 0; i < 2; ++i)
        *(uint4*)&smem[F_B0 + bRow[i] * LDA + bCol[i]] = br[i];
    __syncthreads();

    const int nk = IN_F / BK;
    for (int kt = 0; kt < nk; ++kt) {
        const int buf = kt & 1;
        const unsigned cAoff = buf ? F_A1 : F_A0;
        const unsigned cBoff = buf ? F_B1 : F_B0;

        if (kt + 1 < nk) {
            const int k0 = (kt + 1) * BK;
#pragma unroll
            for (int i = 0; i < 4; ++i)
                ar[i] = *(const float4*)&X[(size_t)(bm + aRow[i]) * IN_F + k0 + aCol[i]];
#pragma unroll
            for (int i = 0; i < 2; ++i)
                br[i] = *(const uint4*)&Wq[(size_t)(bn + bRow[i]) * IN_F + k0 + bCol[i]];
        }

        v16bf af[2], bfr[4];
#pragma unroll
        for (int mt = 0; mt < 2; ++mt) {
            const unsigned base = cAoff + (wm * 32 + mt * 16 + r16) * LDA;
            FragU f;
            f.q[0] = *(const uint4*)&smem[base + kA];
            f.q[1] = *(const uint4*)&smem[base + kA + 16];
            af[mt] = f.v;
        }
#pragma unroll
        for (int nt = 0; nt < 4; ++nt) {
            const unsigned base = cBoff + (wn * 64 + nt * 16 + r16) * LDA + kB;
            FragU f;
            f.q[0] = *(const uint4*)&smem[base];
            f.q[1] = *(const uint4*)&smem[base + 16];
            bfr[nt] = f.v;
        }
#pragma unroll
        for (int mt = 0; mt < 2; ++mt)
#pragma unroll
            for (int nt = 0; nt < 4; ++nt)
                acc[mt][nt] = __builtin_amdgcn_wmma_f32_16x16x32_bf16(
                    false, af[mt], false, bfr[nt], (short)0, acc[mt][nt], false, false);

        if (kt + 1 < nk) {
            const unsigned nAoff = buf ? F_A0 : F_A1;
            const unsigned nBoff = buf ? F_B0 : F_B1;
#pragma unroll
            for (int i = 0; i < 4; ++i) {
                uint2 t; t.x = pack_bf16_2(ar[i].x, ar[i].y); t.y = pack_bf16_2(ar[i].z, ar[i].w);
                *(uint2*)&smem[nAoff + aRow[i] * LDA + aCol[i]] = t;
            }
#pragma unroll
            for (int i = 0; i < 2; ++i)
                *(uint4*)&smem[nBoff + bRow[i] * LDA + bCol[i]] = br[i];
        }
        __syncthreads();
    }

    const int rhi = (lane >> 4) * 8;
#pragma unroll
    for (int mt = 0; mt < 2; ++mt)
#pragma unroll
        for (int nt = 0; nt < 4; ++nt)
#pragma unroll
            for (int r = 0; r < 8; ++r) {
                float v = acc[mt][nt][r];
                v = fminf(50.0f, fmaxf(-50.0f, v));
                const int row = bm + wm * 32 + mt * 16 + rhi + r;
                const int col = bn + wn * 64 + nt * 16 + r16;
                Out[(size_t)row * OUT_F + col] = v;
            }
}

// ---------------------------------------------------------------------------
extern "C" void kernel_launch(void* const* d_in, const int* in_sizes, int n_in,
                              void* d_out, int out_size, void* d_ws, size_t ws_size,
                              hipStream_t stream) {
    const float*          x         = (const float*)d_in[0];
    const int*            indices   = (const int*)d_in[1];
    const unsigned short* codebooks = (const unsigned short*)d_in[2];
    const float*          scales    = (const float*)d_in[3];
    float*                out       = (float*)d_out;

    const int M = in_sizes[0] / IN_F;                 // 8192 tokens
    const size_t wq_bytes = (size_t)OUT_F * IN_F * 2; // 33.5 MB
    const size_t xb_bytes = (size_t)M * IN_F * 2;     // 67 MB

    unsigned short* Wq = (unsigned short*)d_ws;

    aq_dequant_kernel<<<(OUT_F * NGRP) / 256, 256, 0, stream>>>(indices, codebooks, scales, Wq);

    dim3 grid(OUT_F / BN, M / BM);                    // (32, 64)
    if (ws_size >= wq_bytes + xb_bytes) {
        unsigned short* Xb = (unsigned short*)((char*)d_ws + wq_bytes);
        const int xblocks = (M * IN_F) / (256 * 8);   // 8 elts per thread
        aq_xcvt_kernel<<<xblocks, 256, 0, stream>>>(x, Xb);
        aq_gemm_bf16_kernel<<<grid, 256, 0, stream>>>(Xb, Wq, out, M);
    } else {
        aq_gemm_f32_kernel<<<grid, 256, 0, stream>>>(x, Wq, out, M);
    }

    (void)n_in; (void)out_size;
}
